// Atom_57732950393048
// MI455X (gfx1250) — matrix operand order repaired and checked by reading, not compiled, not run on hardware
//
#include <hip/hip_runtime.h>

// CDNA5 / gfx1250 fused kernel: nucleus (GEMM + sin + sigmoid + spline) + RoPE.
// HBM-bound (~85 MB @ 23.3 TB/s): single fused pass, f16 WMMA with f32 accumulate.
// Spline evaluated in O(1) via prefix sums in LDS; freqs pre-converted to f16.

typedef __attribute__((ext_vector_type(16))) _Float16 v16h;
typedef __attribute__((ext_vector_type(8)))  float    v8f;

#define K_DIM      768
#define F_DIM      512
#define SEQ_N      2048
#define NBINS      16
#define LDS_STRIDE 776     // 768 + 8 halves pad -> conflict-free ds_load_b128

// ---------- pre-pass: perm_freqs f32 -> f16 into workspace (768 KB, L2-resident) ----------
__global__ __launch_bounds__(256)
void cvt_freqs_f16(const float* __restrict__ src, _Float16* __restrict__ dst) {
    const int i = (blockIdx.x * 256 + threadIdx.x) * 8;
    const float4* s = reinterpret_cast<const float4*>(src + i);
    float4 f0 = s[0], f1 = s[1];
    union { uint4 u; _Float16 h[8]; } pk;
    pk.h[0] = (_Float16)f0.x; pk.h[1] = (_Float16)f0.y;
    pk.h[2] = (_Float16)f0.z; pk.h[3] = (_Float16)f0.w;
    pk.h[4] = (_Float16)f1.x; pk.h[5] = (_Float16)f1.y;
    pk.h[6] = (_Float16)f1.z; pk.h[7] = (_Float16)f1.w;
    *reinterpret_cast<uint4*>(dst + i) = pk.u;
}

template <bool PRECONV>
__global__ __launch_bounds__(256)
void fused_nucleus_rope(const float*    __restrict__ x,        // (16384, 768)
                        const float*    __restrict__ freqs,    // (512, 768) f32 (fallback)
                        const _Float16* __restrict__ freqs16,  // (512, 768) f16 in ws
                        const float*    __restrict__ phase,    // (512, 1)
                        const float*    __restrict__ heights,  // (512, 16)
                        const float*    __restrict__ bias,     // (512,)
                        const int*      __restrict__ offset_p, // scalar
                        float*          __restrict__ out)      // (16384, 512)
{
    __shared__ __align__(16) _Float16 ldsA[16][LDS_STRIDE];   // 24832 B
    __shared__ float2 splTab[8][16][NBINS + 1];               // 17408 B, padded stride

    const int tid   = threadIdx.x;
    const int lane  = tid & 31;
    const int wave  = tid >> 5;
    const int Mbase = blockIdx.y << 4;                 // 16 tokens per block
    const int Fbase = (blockIdx.x << 7) + (wave << 4); // 128 features/block, 16/wave

    // ---------------- stage x tile (16 x 768 f32 -> f16) into LDS ----------------
    #pragma unroll
    for (int it = 0; it < 6; ++it) {
        int chunk = tid + it * 256;                 // 1536 chunks of 8 floats
        int row   = chunk / 96;
        int c8    = (chunk % 96) * 8;
        const float4* gp = reinterpret_cast<const float4*>(
            x + (size_t)(Mbase + row) * K_DIM + c8);
        float4 f0 = gp[0], f1 = gp[1];
        union { uint4 u; _Float16 h[8]; } pk;
        pk.h[0] = (_Float16)f0.x; pk.h[1] = (_Float16)f0.y;
        pk.h[2] = (_Float16)f0.z; pk.h[3] = (_Float16)f0.w;
        pk.h[4] = (_Float16)f1.x; pk.h[5] = (_Float16)f1.y;
        pk.h[6] = (_Float16)f1.z; pk.h[7] = (_Float16)f1.w;
        *reinterpret_cast<uint4*>(&ldsA[row][c8]) = pk.u;
    }

    // ---------------- per-lane constants (one feature per lane pair-half) ----------------
    const int   f       = Fbase + (lane & 15);
    const int   fi      = lane & 15;
    const float phase_f = phase[f];
    const float bias_f  = bias[f];

    // Spline prefix sums: val(u) = u*W[t] - S[t] + bias, t = floor(15u).
    if (lane < 16) {
        const float4* hp = reinterpret_cast<const float4*>(heights + (size_t)f * NBINS);
        float4 hv0 = hp[0], hv1 = hp[1], hv2 = hp[2], hv3 = hp[3];
        float h[NBINS] = { hv0.x, hv0.y, hv0.z, hv0.w, hv1.x, hv1.y, hv1.z, hv1.w,
                           hv2.x, hv2.y, hv2.z, hv2.w, hv3.x, hv3.y, hv3.z, hv3.w };
        float Wacc = 0.0f, Sacc = 0.0f;
        #pragma unroll
        for (int k = 0; k < NBINS; ++k) {
            float wk = __logf(1.0f + __expf(h[k]));  // softplus
            Wacc += wk;
            Sacc += wk * ((float)k * (1.0f / 15.0f));
            splTab[wave][fi][k] = make_float2(Wacc, Sacc);
        }
    }

    // RoPE inverse frequencies for this lane's pair j = f>>1
    const float e     = (float)(f >> 1) * (1.0f / 256.0f);
    const float inv_h = __expf(-9.210340372f  * e);   // 10000^-e
    const float inv_d = __expf(-11.512925465f * e);   // 100000^-e
    const int   off   = offset_p[0];

    // ---------------- WMMA main loop over K ----------------
    // A frag (16-bit 16x32): lanes 0-15: M=lane, K 0..15;        lanes 16-31: M=lane-16, K 16..31
    // B frag (16-bit 32x16): lanes 0-15: N=lane, K 0..15;        lanes 16-31: N=lane-16, K 16..31
    const int mrow  = lane & 15;
    const int kcolA = (lane >> 4) << 4;   // 0 or 16
    const int krowB = (lane >> 4) << 4;   // 0 or 16
    const _Float16* brow16 = PRECONV ? (freqs16 + (size_t)f * K_DIM + krowB) : nullptr;
    const float*    browf  = PRECONV ? nullptr : (freqs + (size_t)f * K_DIM + krowB);

    v8f acc = {};

    __syncthreads();

    #pragma unroll 4
    for (int kb = 0; kb < K_DIM; kb += 32) {
        // A from LDS (two aligned b128 loads)
        union { uint4 u[2]; v16h h; } a;
        a.u[0] = *reinterpret_cast<const uint4*>(&ldsA[mrow][kb + kcolA]);
        a.u[1] = *reinterpret_cast<const uint4*>(&ldsA[mrow][kb + kcolA + 8]);

        v16h bh;
        if (PRECONV) {
            union { uint4 u[2]; v16h h; } b;
            const uint4* bp = reinterpret_cast<const uint4*>(brow16 + kb);
            b.u[0] = bp[0];
            b.u[1] = bp[1];
            if (kb + 32 < K_DIM) __builtin_prefetch(brow16 + kb + 32, 0, 1);
            bh = b.h;
        } else {
            const float4* bp = reinterpret_cast<const float4*>(browf + kb);
            float4 b0 = bp[0], b1 = bp[1], b2 = bp[2], b3 = bp[3];
            if (kb + 32 < K_DIM) __builtin_prefetch(browf + kb + 32, 0, 1);
            bh[0]  = (_Float16)b0.x; bh[1]  = (_Float16)b0.y;
            bh[2]  = (_Float16)b0.z; bh[3]  = (_Float16)b0.w;
            bh[4]  = (_Float16)b1.x; bh[5]  = (_Float16)b1.y;
            bh[6]  = (_Float16)b1.z; bh[7]  = (_Float16)b1.w;
            bh[8]  = (_Float16)b2.x; bh[9]  = (_Float16)b2.y;
            bh[10] = (_Float16)b2.z; bh[11] = (_Float16)b2.w;
            bh[12] = (_Float16)b3.x; bh[13] = (_Float16)b3.y;
            bh[14] = (_Float16)b3.z; bh[15] = (_Float16)b3.w;
        }

        acc = __builtin_amdgcn_wmma_f32_16x16x32_f16(
            /*neg_a=*/false, a.h, /*neg_b=*/false, bh,
            /*c_mod=*/(short)0, acc, /*reuse_a=*/false, /*reuse_b=*/false);
    }

    // ---------------- fused epilogue: sin -> sigmoid -> spline(O(1)) -> RoPE ----------------
    // C/D layout: VGPR r holds (M = r + 8*(lane>>4), N = lane&15).
    const int rbase = (lane >> 4) << 3;
    const int fodd  = f & 1;

    #pragma unroll
    for (int r = 0; r < 8; ++r) {
        const int   m     = Mbase + rbase + r;                 // flat token row
        const float basis = __sinf(acc[r] + phase_f);
        const float u     = __builtin_amdgcn_rcpf(1.0f + __expf(-basis));  // sigmoid

        // O(1) spline: t = floor(15u) in [0,15], val = u*W[t] - S[t] + bias
        int t = (int)(u * 15.0f);
        t = t > 15 ? 15 : t;
        const float2 ws2 = splTab[wave][fi][t];                // ds_load_b64
        const float  spl = fmaf(u, ws2.x, bias_f - ws2.y);

        // RoPE angle for (token n, pair j)
        const int   ntok  = m & (SEQ_N - 1);
        const float idx   = (float)(ntok + off);
        const float days  = floorf(idx * (1.0f / 64.0f));
        const float hours = idx - days * 64.0f;
        const float ang   = hours * inv_h + days * inv_d;
        const float cs    = __cosf(ang);
        const float sn    = __sinf(ang);

        // partner value lives in the adjacent lane (feature parity == lane parity)
        const float other = __shfl_xor(spl, 1, 32);
        const float o = fodd ? fmaf(other, sn,  spl * cs)      // ro = xe*sin + xo*cos
                             : fmaf(spl,  cs, -(other * sn));  // re = xe*cos - xo*sin

        out[(size_t)m * F_DIM + f] = o;
    }
}

extern "C" void kernel_launch(void* const* d_in, const int* in_sizes, int n_in,
                              void* d_out, int out_size, void* d_ws, size_t ws_size,
                              hipStream_t stream) {
    (void)in_sizes; (void)n_in; (void)out_size;
    const float* x       = (const float*)d_in[0];
    const float* freqs   = (const float*)d_in[1];
    const float* phase   = (const float*)d_in[2];
    const float* heights = (const float*)d_in[3];
    const float* bias    = (const float*)d_in[4];
    const int*   offset  = (const int*)  d_in[5];
    float*       out     = (float*)d_out;

    const size_t needB = (size_t)F_DIM * K_DIM * sizeof(_Float16);  // 768 KB
    dim3 grid(F_DIM / 128, (8 * SEQ_N) / 16);  // (4, 1024)
    dim3 block(256);                            // 8 waves (wave32)

    if (ws_size >= needB) {
        _Float16* f16w = (_Float16*)d_ws;
        cvt_freqs_f16<<<(F_DIM * K_DIM) / (256 * 8), 256, 0, stream>>>(freqs, f16w);
        fused_nucleus_rope<true><<<grid, block, 0, stream>>>(
            x, freqs, f16w, phase, heights, bias, offset, out);
    } else {
        fused_nucleus_rope<false><<<grid, block, 0, stream>>>(
            x, freqs, nullptr, phase, heights, bias, offset, out);
    }
}
